// ConvOffset3d_53721450939001
// MI455X (gfx1250) — matrix operand order repaired and checked by reading, not compiled, not run on hardware
//
#include <hip/hip_runtime.h>

typedef __attribute__((ext_vector_type(16))) _Float16 v16h;
typedef __attribute__((ext_vector_type(8)))  _Float16 v8h;
typedef __attribute__((ext_vector_type(4)))  _Float16 v4h;
typedef __attribute__((ext_vector_type(8)))  float    v8f;

// Problem constants (from reference): B=2, C=64, T=8, H=32, W=32, Co=64, 3x3x3
#define Tdim 8
#define Hdim 32
#define Wdim 32
#define THW  8192            // T*H*W
#define Cch  64
#define Cout 64
#define Ktap 27
#define CK   1728            // Cch * Ktap
#define Ldim 8192            // To*Ho*Wo (stride 1, pad 1)
#define NT   32              // output voxels per workgroup
#define CROW 72              // padded LDS row (64 ch + 8 pad halves, 144B)

// ---------------- weight f32 -> f16, reordered r = k*64 + c ----------------
__global__ __launch_bounds__(256)
void wcvt_kernel(const float* __restrict__ w, _Float16* __restrict__ wh) {
    int idx = blockIdx.x * blockDim.x + threadIdx.x;
    if (idx >= Cout * CK) return;
    int co = idx / CK;
    int r  = idx - co * CK;
    int k  = r >> 6;          // r / 64
    int c  = r & 63;          // r % 64
    // weight layout (Co, C, kT,kH,kW): element = w[(co*64 + c)*27 + k]
    wh[idx] = (_Float16)w[(co * Cch + c) * Ktap + k];
}

// ---------------- fused im2col (trilinear) + WMMA GEMM ----------------
__global__ __launch_bounds__(256)
void ConvOffset3d_wmma_kernel(const float* __restrict__ x,
                              const float* __restrict__ off,
                              const _Float16* __restrict__ wh,
                              const float* __restrict__ bias,
                              float* __restrict__ out) {
    __shared__ float    s_pos[Ktap][NT][3];       // deformed sample positions
    __shared__ _Float16 s_colsT[NT][CROW];        // [voxel][channel] f16 tile

    const int lblk   = blockIdx.x & (Ldim / NT - 1);   // 256 l-tiles
    const int b      = blockIdx.x / (Ldim / NT);
    const int l_base = lblk * NT;

    const int lane   = threadIdx.x & 31;
    const int wave   = threadIdx.x >> 5;        // 0..7
    const int coTile = wave & 3;                // 4 tiles of 16 output channels
    const int lTile  = wave >> 2;               // 2 tiles of 16 voxels

    // ---- Stage A: deformed sampling positions for all (tap, voxel) ----
    for (int t = threadIdx.x; t < Ktap * NT; t += blockDim.x) {
        int k  = t / NT;
        int li = t - k * NT;
        int l  = l_base + li;
        int to = l >> 10, ho = (l >> 5) & 31, wo = l & 31;
        int kt = k / 9, kh = (k / 3) % 3, kw = k % 3;
        const float* op = off + ((size_t)b * 81 + (size_t)k * 3) * (size_t)Ldim + l;
        s_pos[k][li][0] = (float)(to + kt - 1) + op[0];
        s_pos[k][li][1] = (float)(ho + kh - 1) + op[(size_t)Ldim];
        s_pos[k][li][2] = (float)(wo + kw - 1) + op[2 * (size_t)Ldim];
    }

    v8f acc = {};

    // per-wave constant fragment addressing
    const int m   = lane & 15;                  // row within A tile / col within B,D
    const int klo = (lane < 16) ? 0 : 8;        // A split-K half
    const int kb  = (lane < 16) ? 0 : 16;       // unused; B fragments split K by klo too
    const int liB = lTile * 16 + m;             // voxel column for B fragment

    // ---- main loop over 27 taps; 64-channel column tile per tap ----
    for (int k = 0; k < Ktap; ++k) {
        __syncthreads();   // previous tile fully consumed

        // build 32(voxel) x 64(channel) f16 column tile, corners once per tap
        {
            const int li = threadIdx.x & 31;
            const int cg = threadIdx.x >> 5;    // 0..7 -> 8 channels each
            const float pt = s_pos[k][li][0];
            const float ph = s_pos[k][li][1];
            const float pw = s_pos[k][li][2];
            const float t0 = floorf(pt), h0 = floorf(ph), w0 = floorf(pw);
            const float ft = pt - t0, fh = ph - h0, fw = pw - w0;
            const int it = (int)t0, ih = (int)h0, iw = (int)w0;

            int   lin[8];
            float wgt[8];
#pragma unroll
            for (int cn = 0; cn < 8; ++cn) {
                const int ct = cn >> 2, ch = (cn >> 1) & 1, cw = cn & 1;
                const int ti = it + ct, hi = ih + ch, wi = iw + cw;
                const bool valid = (ti >= 0) & (ti < Tdim) & (hi >= 0) &
                                   (hi < Hdim) & (wi >= 0) & (wi < Wdim);
                const float wv = (ct ? ft : 1.f - ft) * (ch ? fh : 1.f - fh) *
                                 (cw ? fw : 1.f - fw);
                const int tic = min(max(ti, 0), Tdim - 1);
                const int hic = min(max(hi, 0), Hdim - 1);
                const int wic = min(max(wi, 0), Wdim - 1);
                lin[cn] = (tic * Hdim + hic) * Wdim + wic;
                wgt[cn] = valid ? wv : 0.f;
            }
            const float* xb = x + ((size_t)b * Cch + cg * 8) * (size_t)THW;
            v4h pk0, pk1;
#pragma unroll
            for (int j = 0; j < 8; ++j) {
                const float* xc = xb + (size_t)j * THW;
                float v = 0.f;
#pragma unroll
                for (int cn = 0; cn < 8; ++cn) v += xc[lin[cn]] * wgt[cn];
                if (j < 4) pk0[j] = (_Float16)v;
                else       pk1[j - 4] = (_Float16)v;
            }
            // two aligned 8B stores -> ds_store_b64
            *(v4h*)&s_colsT[li][cg * 8]     = pk0;
            *(v4h*)&s_colsT[li][cg * 8 + 4] = pk1;
        }
        __syncthreads();

        // ---- A fragments: 16x32 f16 weights, two K-chunks of this tap ----
        const _Float16* wrow =
            wh + (size_t)(coTile * 16 + m) * CK + (size_t)k * 64;
        v16h a0, a1;
#pragma unroll
        for (int i = 0; i < 8; ++i) {
            a0[i]     = wrow[klo + i];
            a0[8 + i] = wrow[16 + klo + i];
            a1[i]     = wrow[32 + klo + i];
            a1[8 + i] = wrow[48 + klo + i];
        }

        // ---- B fragments: two 8-half runs (k = klo.., 16+klo..) -> ds_load_b128 x2 each ----
        const v8h b0lo = *(const v8h*)&s_colsT[liB][klo];
        const v8h b0hi = *(const v8h*)&s_colsT[liB][16 + klo];
        const v8h b1lo = *(const v8h*)&s_colsT[liB][32 + klo];
        const v8h b1hi = *(const v8h*)&s_colsT[liB][48 + klo];
        v16h b0, b1;
#pragma unroll
        for (int i = 0; i < 8; ++i) {
            b0[i] = b0lo[i]; b0[8 + i] = b0hi[i];
            b1[i] = b1lo[i]; b1[8 + i] = b1hi[i];
        }

        acc = __builtin_amdgcn_wmma_f32_16x16x32_f16(
            false, a0, false, b0, (short)0, acc, false, false);
        acc = __builtin_amdgcn_wmma_f32_16x16x32_f16(
            false, a1, false, b1, (short)0, acc, false, false);
    }

    // ---- epilogue: bias + store f32 output ----
    const int lo = l_base + lTile * 16 + m;
#pragma unroll
    for (int r = 0; r < 8; ++r) {
        const int co = coTile * 16 + ((lane < 16) ? r : r + 8);
        out[((size_t)b * Cout + co) * (size_t)Ldim + lo] = acc[r] + bias[co];
    }
}

extern "C" void kernel_launch(void* const* d_in, const int* in_sizes, int n_in,
                              void* d_out, int out_size, void* d_ws, size_t ws_size,
                              hipStream_t stream) {
    const float* x    = (const float*)d_in[0];
    const float* off  = (const float*)d_in[1];
    const float* w    = (const float*)d_in[2];
    const float* bias = (const float*)d_in[3];
    float* out        = (float*)d_out;

    _Float16* wh = (_Float16*)d_ws;   // 64*1728*2 = 221,184 bytes

    wcvt_kernel<<<(Cout * CK + 255) / 256, 256, 0, stream>>>(w, wh);

    const int nblocks = 2 * (Ldim / NT);   // B * 256 = 512 workgroups
    ConvOffset3d_wmma_kernel<<<nblocks, 256, 0, stream>>>(x, off, wh, bias, out);
}
